// GAT_64003602645176
// MI455X (gfx1250) — matrix operand order repaired and checked by reading, not compiled, not run on hardware
//
#include <hip/hip_runtime.h>

// ---------------------------------------------------------------------------
// GAT forward for MI455X (gfx1250, wave32).
// GEMMs: v_wmma_f32_16x16x32_f16 with pre-packed f16 weight fragments and
// dual accumulator chains.  Edge softmax/aggregation: native float atomics.
// ---------------------------------------------------------------------------

typedef __attribute__((ext_vector_type(16))) _Float16 v16h;
typedef __attribute__((ext_vector_type(8)))  float    v8f;

#define FEATW 64   // all GEMM outputs are 64 wide
#define NEG_SLOPE 0.2f

// ---------------------------------------------------------------------------
// Pack W[K x 64] (f32, row-major) into f16 WMMA B-fragment order:
//   Wp[((kt*4 + ct)*512) + lane*16 + i] = W[(kt*32 + 16*(lane>>4) + i)*64
//                                          + ct*16 + (lane&15)]
// so each lane of the GEMM reads its 16-element fragment contiguously (32B).
// ---------------------------------------------------------------------------
__global__ void k_pack_w(const float* __restrict__ W, _Float16* __restrict__ Wp, int K) {
  int p = blockIdx.x * 256 + threadIdx.x;
  if (p >= K * FEATW) return;
  int i    = p & 15;
  int lane = (p >> 4) & 31;
  int tile = p >> 9;               // kt*4 + ct
  int ct   = tile & 3;
  int kt   = tile >> 2;
  int k = kt * 32 + 16 * (lane >> 4) + i;
  int n = ct * 16 + (lane & 15);
  Wp[p] = (_Float16)W[(size_t)k * FEATW + n];
}

// A fragment: element i <-> K = kb + 16*half + i  (matches packed B order)
__device__ __forceinline__ v16h load_a_frag(const float* __restrict__ A,
                                            size_t rowOff, int kb, int half, int reluA) {
  const float4* p0 = reinterpret_cast<const float4*>(A + rowOff + kb + half * 16);
  float4 s0 = p0[0];
  float4 s1 = p0[1];
  const float4* p1 = reinterpret_cast<const float4*>(A + rowOff + kb + half * 16 + 8);
  float4 s2 = p1[0];
  float4 s3 = p1[1];
  float tmp[16] = { s0.x, s0.y, s0.z, s0.w, s1.x, s1.y, s1.z, s1.w,
                    s2.x, s2.y, s2.z, s2.w, s3.x, s3.y, s3.z, s3.w };
  if (reluA) {
    #pragma unroll
    for (int i = 0; i < 16; ++i) tmp[i] = fmaxf(tmp[i], 0.0f);
  }
  v16h a;
  #pragma unroll
  for (int i = 0; i < 16; ++i) a[i] = (_Float16)tmp[i];
  return a;
}

__device__ __forceinline__ v16h load_b_frag(const _Float16* __restrict__ Wp,
                                            int kb, int colTile, int lane) {
  return *reinterpret_cast<const v16h*>(Wp + ((size_t)(kb >> 5) * 4 + colTile) * 512 + lane * 16);
}

// ---------------------------------------------------------------------------
// C[M x 64] = act(A)[M x K] @ Wp (+ bias[64]).  One wave per 16x16 tile,
// 8 tiles per 256-thread block.  K is a multiple of 64 (256 or 64), split
// into two independent WMMA accumulator chains for ILP on the matrix pipe.
// ---------------------------------------------------------------------------
__global__ __launch_bounds__(256) void k_gemm_wmma(
    const float* __restrict__ A, const _Float16* __restrict__ Wp,
    const float* __restrict__ bias, float* __restrict__ C,
    int M, int K, int reluA)
{
  const int lane = threadIdx.x & 31;
  const int wave = threadIdx.x >> 5;
  const int tile = blockIdx.x * 8 + wave;
  const int rowTile = tile >> 2;          // 4 col tiles (64 wide)
  const int colTile = tile & 3;
  if (rowTile * 16 >= M) return;

  const int half = lane >> 4;             // 0: lanes 0-15, 1: lanes 16-31
  const int mr   = rowTile * 16 + (lane & 15);   // A row for this lane
  const int nc   = colTile * 16 + (lane & 15);   // B/D column for this lane
  const size_t rowOff = (size_t)mr * K;

  v8f acc0 = {}, acc1 = {};
  if (bias) {
    const float bv = bias[nc];
    #pragma unroll
    for (int r = 0; r < 8; ++r) acc0[r] = bv;
  }

  for (int kb = 0; kb < K; kb += 64) {
    if (kb + 64 < K) __builtin_prefetch(A + rowOff + kb + 64, 0, 0);
    v16h a0 = load_a_frag(A, rowOff, kb,      half, reluA);
    v16h b0 = load_b_frag(Wp, kb,      colTile, lane);
    v16h a1 = load_a_frag(A, rowOff, kb + 32, half, reluA);
    v16h b1 = load_b_frag(Wp, kb + 32, colTile, lane);
    acc0 = __builtin_amdgcn_wmma_f32_16x16x32_f16(false, a0, false, b0,
                                                  (short)0, acc0, false, false);
    acc1 = __builtin_amdgcn_wmma_f32_16x16x32_f16(false, a1, false, b1,
                                                  (short)0, acc1, false, false);
  }

  // D layout: reg r -> row = rowTile*16 + 8*half + r, col = nc
  float* cp = C + (size_t)(rowTile * 16 + half * 8) * FEATW + nc;
  #pragma unroll
  for (int r = 0; r < 8; ++r) cp[r * FEATW] = acc0[r] + acc1[r];
}

// ---------------------------------------------------------------------------
// Per-node attention scores: s[n] = Hc[n,:] . a_s ; d[n] = Hc[n,:] . a_d
// One wave per node (64 feats -> 2 per lane), wave32 shuffle reduction.
// ---------------------------------------------------------------------------
__global__ __launch_bounds__(256) void k_node_scores(
    const float* __restrict__ Hc, const float* __restrict__ as_,
    const float* __restrict__ ad_, float* __restrict__ sOut,
    float* __restrict__ dOut, int N)
{
  const int lane = threadIdx.x & 31;
  const int node = blockIdx.x * 8 + (threadIdx.x >> 5);
  if (node >= N) return;
  float2 h  = *reinterpret_cast<const float2*>(Hc + (size_t)node * FEATW + lane * 2);
  float2 va = *reinterpret_cast<const float2*>(as_ + lane * 2);
  float2 vb = *reinterpret_cast<const float2*>(ad_ + lane * 2);
  float ps = h.x * va.x + h.y * va.y;
  float pd = h.x * vb.x + h.y * vb.y;
  #pragma unroll
  for (int off = 16; off > 0; off >>= 1) {
    ps += __shfl_down(ps, off, 32);
    pd += __shfl_down(pd, off, 32);
  }
  if (lane == 0) { sOut[node] = ps; dOut[node] = pd; }
}

__global__ void k_fill(float* __restrict__ p, float v, int n) {
  int i = blockIdx.x * 256 + threadIdx.x;
  if (i < n) p[i] = v;
}

// Order-preserving float atomic max (handles mixed signs; init = -3e38).
__device__ __forceinline__ void atomicMaxF(float* addr, float val) {
  if (__float_as_int(val) >= 0)
    atomicMax((int*)addr, __float_as_int(val));
  else
    atomicMin((unsigned int*)addr, __float_as_uint(val));
}

// Edge logits + segment max.  Edges [0,E) from src/dst, [E,ET) are self-loops.
__global__ void k_edge_logits(
    const int* __restrict__ src, const int* __restrict__ dst,
    const float* __restrict__ sArr, const float* __restrict__ dArr,
    float* __restrict__ eOut, float* __restrict__ mArr, int E, int ET)
{
  int g = blockIdx.x * 256 + threadIdx.x;
  if (g >= ET) return;
  int sn = (g < E) ? src[g] : (g - E);
  int dn = (g < E) ? dst[g] : (g - E);
  float x = sArr[sn] + dArr[dn];
  float e = (x > 0.0f) ? x : NEG_SLOPE * x;
  eOut[g] = e;
  atomicMaxF(&mArr[dn], e);
}

// ex = exp(e - m[dst]); den[dst] += ex  (in-place over the logit array)
__global__ void k_edge_exp(
    const int* __restrict__ dst, const float* __restrict__ mArr,
    float* __restrict__ eArr, float* __restrict__ den, int E, int ET)
{
  int g = blockIdx.x * 256 + threadIdx.x;
  if (g >= ET) return;
  int dn = (g < E) ? dst[g] : (g - E);
  float ex = __expf(eArr[g] - mArr[dn]);
  eArr[g] = ex;
  atomicAdd(&den[dn], ex);
}

// out[n, f] = bias[f]  (aggregation accumulator init)
__global__ void k_init_out(float* __restrict__ out, const float* __restrict__ b, int total) {
  int i = blockIdx.x * 256 + threadIdx.x;
  if (i < total) out[i] = b[i & (FEATW - 1)];
}

// out[dst] += (ex/den[dst]) * Hc[src] ; 64 threads per edge (feature = tid%64)
__global__ __launch_bounds__(256) void k_aggregate(
    const int* __restrict__ src, const int* __restrict__ dst,
    const float* __restrict__ ex, const float* __restrict__ den,
    const float* __restrict__ Hc, float* __restrict__ out, int E, int ET)
{
  int g = blockIdx.x * 4 + (threadIdx.x >> 6);
  int f = threadIdx.x & (FEATW - 1);
  if (g >= ET) return;
  int sn = (g < E) ? src[g] : (g - E);
  int dn = (g < E) ? dst[g] : (g - E);
  float alpha = ex[g] / (den[dn] + 1e-16f);
  atomicAdd(&out[(size_t)dn * FEATW + f], alpha * Hc[(size_t)sn * FEATW + f]);
}

// Row-wise L2 normalize: one wave per node.
__global__ __launch_bounds__(256) void k_normalize(
    const float* __restrict__ h, float* __restrict__ out, int N)
{
  const int lane = threadIdx.x & 31;
  const int node = blockIdx.x * 8 + (threadIdx.x >> 5);
  if (node >= N) return;
  float2 v = *reinterpret_cast<const float2*>(h + (size_t)node * FEATW + lane * 2);
  float ss = v.x * v.x + v.y * v.y;
  #pragma unroll
  for (int off = 16; off > 0; off >>= 1) ss += __shfl_xor(ss, off, 32);
  float inv = 1.0f / fmaxf(sqrtf(ss), 1e-12f);
  float2 r; r.x = v.x * inv; r.y = v.y * inv;
  *reinterpret_cast<float2*>(out + (size_t)node * FEATW + lane * 2) = r;
}

// ---------------------------------------------------------------------------
// Host orchestration
// ---------------------------------------------------------------------------
extern "C" void kernel_launch(void* const* d_in, const int* in_sizes, int n_in,
                              void* d_out, int out_size, void* d_ws, size_t ws_size,
                              hipStream_t stream) {
  const float* x     = (const float*)d_in[0];
  const int*   src   = (const int*)  d_in[1];
  const int*   dst   = (const int*)  d_in[2];
  const float* lin_w = (const float*)d_in[3];
  const float* lin_b = (const float*)d_in[4];
  const float* Wl[3]  = { (const float*)d_in[5], (const float*)d_in[9],  (const float*)d_in[13] };
  const float* Asl[3] = { (const float*)d_in[6], (const float*)d_in[10], (const float*)d_in[14] };
  const float* Adl[3] = { (const float*)d_in[7], (const float*)d_in[11], (const float*)d_in[15] };
  const float* Bl[3]  = { (const float*)d_in[8], (const float*)d_in[12], (const float*)d_in[16] };

  const int IN_D = 256;
  const int N  = in_sizes[0] / IN_D;
  const int E  = in_sizes[1];
  const int ET = E + N;                    // with self-loops

  // Workspace layout: Wp (f16, 32KB, 32B-aligned at base) | f32 arrays
  _Float16* Wp = (_Float16*)d_ws;                    // max 256*64 f16
  float* H   = (float*)((char*)d_ws + 32768);
  float* Hc  = H   + (size_t)N * FEATW;
  float* Acc = Hc  + (size_t)N * FEATW;
  float* sA  = Acc + (size_t)N * FEATW;
  float* dA  = sA  + N;
  float* mA  = dA  + N;
  float* den = mA  + N;
  float* ex  = den + N;

  auto gemmGrid = [](int M) { return (((M + 15) / 16) * 4 + 7) / 8; };
  const int nodeGrid = (N + 7) / 8;
  const int edgeGrid = (ET + 255) / 256;
  const int nGrid    = (N + 255) / 256;
  const int nfGrid   = (N * FEATW + 255) / 256;
  const int aggGrid  = (ET + 3) / 4;

  // Feature linear: H = x @ lin_w + lin_b
  k_pack_w<<<(IN_D * FEATW + 255) / 256, 256, 0, stream>>>(lin_w, Wp, IN_D);
  k_gemm_wmma<<<gemmGrid(N), 256, 0, stream>>>(x, Wp, lin_b, H, N, IN_D, 0);

  float* cur = H;                          // layer input
  float* nxt[3] = { Acc, H, Acc };         // layer outputs (ping-pong)
  for (int L = 0; L < 3; ++L) {
    const int reluA = (L > 0) ? 1 : 0;     // fold inter-layer ReLU into GEMM A-load
    k_pack_w<<<(FEATW * FEATW + 255) / 256, 256, 0, stream>>>(Wl[L], Wp, FEATW);
    k_gemm_wmma<<<gemmGrid(N), 256, 0, stream>>>(cur, Wp, nullptr, Hc, N, FEATW, reluA);
    k_node_scores<<<nodeGrid, 256, 0, stream>>>(Hc, Asl[L], Adl[L], sA, dA, N);
    k_fill<<<nGrid, 256, 0, stream>>>(mA, -3.0e38f, N);
    k_fill<<<nGrid, 256, 0, stream>>>(den, 0.0f, N);
    k_edge_logits<<<edgeGrid, 256, 0, stream>>>(src, dst, sA, dA, ex, mA, E, ET);
    k_edge_exp<<<edgeGrid, 256, 0, stream>>>(dst, mA, ex, den, E, ET);
    k_init_out<<<nfGrid, 256, 0, stream>>>(nxt[L], Bl[L], N * FEATW);
    k_aggregate<<<aggGrid, 256, 0, stream>>>(src, dst, ex, den, Hc, nxt[L], E, ET);
    cur = nxt[L];
  }

  k_normalize<<<nodeGrid, 256, 0, stream>>>(cur, (float*)d_out, N);
}
